// CSAMBlock_86526411145218
// MI455X (gfx1250) — matrix operand order repaired and checked by reading, not compiled, not run on hardware
//
#include <hip/hip_runtime.h>
#include <hip/hip_bf16.h>
#include <math.h>

// ---------------------------------------------------------------------------
// Swin block on MI455X (gfx1250): all matmuls via v_wmma_f32_16x16x32_f16.
// Windows padded 49 -> 64 rows so every GEMM dim is WMMA-tileable.
// ---------------------------------------------------------------------------

typedef _Float16 h16;
typedef __attribute__((ext_vector_type(8)))  _Float16 v8h;
typedef __attribute__((ext_vector_type(16))) _Float16 v16h;
typedef __attribute__((ext_vector_type(8)))  float    v8f;

#define B_    32
#define H_    56
#define W_    56
#define C_    192
#define NH_   6
#define WS_   7
#define SS_   3
#define NPAD  64          // padded tokens per window
#define NWIN  2048        // B * 8 * 8
#define MWIN  (NWIN*NPAD) // 131072 rows in windowed space
#define MTOK  (B_*H_*W_)  // 100352 rows in token space
#define HID_  768
#define SCALE_ 0.17677669529663687f  // 32^-0.5

static __device__ __forceinline__ v16h cat16(v8h lo, v8h hi) {
  return __builtin_shufflevector(lo, hi, 0,1,2,3,4,5,6,7,8,9,10,11,12,13,14,15);
}
static __device__ __forceinline__ v8f wmma_f16(v16h a, v16h b, v8f c) {
  return __builtin_amdgcn_wmma_f32_16x16x32_f16(false, a, false, b, (short)0, c,
                                                false, false);
}

// ---------------------------------------------------------------------------
// Kernel 1: LN1 + cyclic shift + window partition + f32->f16, zero-pad rows.
// One block (8 waves) per window; each wave handles rows w, w+8, ...
// ---------------------------------------------------------------------------
__global__ __launch_bounds__(256)
void ln1_window_kernel(const float* __restrict__ x,
                       const float* __restrict__ g1,
                       const float* __restrict__ b1,
                       h16* __restrict__ win) {
  const int tid  = threadIdx.x;
  const int lane = tid & 31;
  const int wave = tid >> 5;
  const int w    = blockIdx.x;          // window id
  const int b    = w >> 6;              // image
  const int wi   = w & 63;              // window-in-image
  const int wh   = wi >> 3, ww = wi & 7;

  for (int row = wave; row < NPAD; row += 8) {
    h16* dst = win + ((size_t)w * NPAD + row) * C_;
    if (row < WS_ * WS_) {
      const int i = row / WS_, j = row % WS_;
      const int hh = (wh * WS_ + i + SS_) % H_;   // inverse of roll(-SS)
      const int wc = (ww * WS_ + j + SS_) % W_;
      const float* src = x + ((size_t)b * (H_ * W_) + hh * W_ + wc) * C_;
      float f[6];
      float s = 0.f;
      #pragma unroll
      for (int t = 0; t < 6; ++t) { f[t] = src[lane + 32 * t]; s += f[t]; }
      #pragma unroll
      for (int off = 16; off; off >>= 1) s += __shfl_xor(s, off, 32);
      const float mean = s * (1.f / C_);
      float v = 0.f;
      #pragma unroll
      for (int t = 0; t < 6; ++t) { float d = f[t] - mean; v += d * d; }
      #pragma unroll
      for (int off = 16; off; off >>= 1) v += __shfl_xor(v, off, 32);
      const float rs = rsqrtf(v * (1.f / C_) + 1e-5f);
      #pragma unroll
      for (int t = 0; t < 6; ++t) {
        const int c = lane + 32 * t;
        dst[c] = (h16)((f[t] - mean) * rs * g1[c] + b1[c]);
      }
    } else {
      #pragma unroll
      for (int t = 0; t < 6; ++t) dst[lane + 32 * t] = (h16)0.f;
    }
  }
}

// ---------------------------------------------------------------------------
// Tiled WMMA GEMM: out[M,N] = A[M,K](f16) * W[N,K](f32->f16)^T + bias.
// Block tile 128x64, 8 waves of 32x32 (2x2 WMMA tiles), BK=32 via LDS.
// MODE: 0 = store f16; 1 = store f32; 2 = GELU -> f16; 3 = +res -> f32.
// ---------------------------------------------------------------------------
#define BM 128
#define BN 64
#define BK 32
#define LDT 48   // LDS row pitch in halves (96B: 16B-aligned, staggers banks)

template<int MODE>
__global__ __launch_bounds__(256)
void gemm_wmma(const h16*  __restrict__ A,
               const float* __restrict__ Wt,   // [N,K] row-major
               const float* __restrict__ bias,
               void*        __restrict__ outv,
               const float* __restrict__ res,
               int M, int N, int K) {
  __shared__ __align__(16) h16 As[BM * LDT];
  __shared__ __align__(16) h16 Bs[BN * LDT];

  const int tid  = threadIdx.x;
  const int lane = tid & 31, wave = tid >> 5;
  const int ln   = lane & 15, g = lane >> 4;
  const int wm   = wave >> 1, wn = wave & 1;
  const int m0   = blockIdx.x * BM;
  const int n0   = blockIdx.y * BN;

  v8f acc[2][2];
  #pragma unroll
  for (int i = 0; i < 2; ++i)
    #pragma unroll
    for (int j = 0; j < 2; ++j) acc[i][j] = (v8f){0,0,0,0,0,0,0,0};

  const int arow = tid >> 1, achunk = (tid & 1) * 16;  // 16 halves per thread
  const int brow = tid >> 2, bchunk = (tid & 3) * 8;   // 8 floats per thread

  for (int k0 = 0; k0 < K; k0 += BK) {
    // ---- stage A tile (f16, 128x32) ----
    const h16* ag = A + (size_t)(m0 + arow) * K + k0 + achunk;
    uint4 a0 = *(const uint4*)ag;
    uint4 a1 = *(const uint4*)(ag + 8);
    *(uint4*)&As[arow * LDT + achunk]     = a0;
    *(uint4*)&As[arow * LDT + achunk + 8] = a1;
    if (k0 + BK < K) __builtin_prefetch(ag + BK, 0, 1);  // global_prefetch_b8
    // ---- stage B tile (f32 -> f16, 64x32) ----
    const float* wg = Wt + (size_t)(n0 + brow) * K + k0 + bchunk;
    float4 f0 = *(const float4*)wg;
    float4 f1 = *(const float4*)(wg + 4);
    v8h bh;
    bh[0]=(h16)f0.x; bh[1]=(h16)f0.y; bh[2]=(h16)f0.z; bh[3]=(h16)f0.w;
    bh[4]=(h16)f1.x; bh[5]=(h16)f1.y; bh[6]=(h16)f1.z; bh[7]=(h16)f1.w;
    *(v8h*)&Bs[brow * LDT + bchunk] = bh;
    __syncthreads();

    // ---- fragments (ISA 7.12.2 layouts) ----
    v16h afr[2], bfr[2];
    #pragma unroll
    for (int mt = 0; mt < 2; ++mt) {
      const h16* p = &As[(wm * 32 + mt * 16 + ln) * LDT + g * 8];
      afr[mt] = cat16(*(const v8h*)p, *(const v8h*)(p + 16));
    }
    #pragma unroll
    for (int nt = 0; nt < 2; ++nt) {
      const h16* p = &Bs[(wn * 32 + nt * 16 + ln) * LDT + g * 8];
      bfr[nt] = cat16(*(const v8h*)p, *(const v8h*)(p + 16));
    }
    #pragma unroll
    for (int mt = 0; mt < 2; ++mt)
      #pragma unroll
      for (int nt = 0; nt < 2; ++nt)
        acc[mt][nt] = wmma_f16(afr[mt], bfr[nt], acc[mt][nt]);
    __syncthreads();
  }

  // ---- epilogue ----
  #pragma unroll
  for (int mt = 0; mt < 2; ++mt)
    #pragma unroll
    for (int nt = 0; nt < 2; ++nt) {
      const int col = n0 + wn * 32 + nt * 16 + ln;
      const float bv = bias[col];
      #pragma unroll
      for (int r = 0; r < 8; ++r) {
        const int row = m0 + wm * 32 + mt * 16 + g * 8 + r;
        const size_t idx = (size_t)row * N + col;
        float v = acc[mt][nt][r] + bv;
        if constexpr (MODE == 0) {
          ((h16*)outv)[idx] = (h16)v;
        } else if constexpr (MODE == 1) {
          ((float*)outv)[idx] = v;
        } else if constexpr (MODE == 2) {
          ((h16*)outv)[idx] = (h16)(0.5f * v * (1.f + erff(v * 0.70710678f)));
        } else {
          ((float*)outv)[idx] = v + res[idx];
        }
      }
    }
  (void)res;
}

// ---------------------------------------------------------------------------
// Kernel 3: windowed attention. One block (4 waves) per WINDOW, looping over
// the 6 heads so the 49x49 mask is staged into LDS once per block.
// Per head & wave: S = Q*K^T (4 WMMA), branchless masked softmax (mask read
// from LDS with clamped indices), P via LDS re-layout, O = P*V (4 WMMA)
// against an LDS-transposed V tile staged with coalesced b128 loads.
// ---------------------------------------------------------------------------
__global__ __launch_bounds__(128)
void attn_wmma_kernel(const h16*  __restrict__ qkv,   // [MWIN, 576]
                      const float* __restrict__ mask, // [64,49,49]
                      h16*        __restrict__ out) { // [MWIN, 192]
  __shared__ __align__(16) h16   Vt[32 * 72];      // V transposed [hd][tok]
  __shared__ __align__(16) h16   P[4 * 16 * 72];   // probs per wave [row][tok]
  __shared__ __align__(16) float MS[49 * 49];      // this window's attn mask

  const int tid  = threadIdx.x;
  const int lane = tid & 31, wave = tid >> 5;
  const int ln   = lane & 15, g = lane >> 4;
  const int win  = blockIdx.x;
  const int wi   = win & 63;
  const h16* base = qkv + (size_t)win * NPAD * 576;

  // ---- stage attention mask once per block (coalesced) ----
  for (int i = tid; i < 49 * 49; i += 128)
    MS[i] = mask[(size_t)wi * 2401 + i];

  const int vrow = tid >> 1;            // token 0..63
  const int vc0  = (tid & 1) * 16;      // head-dim chunk 0 / 16

  for (int head = 0; head < NH_; ++head) {
    const int hd0 = head * 32;
    __syncthreads();  // mask staged (iter 0) / previous head done with Vt

    // ---- stage V^T: coalesced b128 row loads, b16 scatter to LDS ----
    {
      const h16* vp = base + (size_t)vrow * 576 + 384 + hd0 + vc0;
      v8h v0 = *(const v8h*)vp;
      v8h v1 = *(const v8h*)(vp + 8);
      #pragma unroll
      for (int j = 0; j < 8; ++j) {
        Vt[(vc0 + j) * 72 + vrow]     = v0[j];
        Vt[(vc0 + 8 + j) * 72 + vrow] = v1[j];
      }
    }

    // ---- Q fragment (A layout: K contiguous innermost) ----
    v16h aQ;
    {
      const h16* p = base + (size_t)(wave * 16 + ln) * 576 + hd0 + g * 8;
      aQ = cat16(*(const v8h*)p, *(const v8h*)(p + 16));
    }

    // ---- S = Q * K^T : 4 WMMA (head-dim 32 == one K-step) ----
    v8f S[4];
    #pragma unroll
    for (int t = 0; t < 4; ++t) {
      const h16* p = base + (size_t)(t * 16 + ln) * 576 + 192 + hd0 + g * 8;
      v16h bK = cat16(*(const v8h*)p, *(const v8h*)(p + 16));
      v8f z = (v8f){0,0,0,0,0,0,0,0};
      S[t] = wmma_f16(aQ, bK, z);
    }

    // ---- branchless masked softmax over 64 cols (cols>=49 -> -1e30) ----
    float mx[8], sm[8];
    #pragma unroll
    for (int r = 0; r < 8; ++r) { mx[r] = -1e30f; sm[r] = 0.f; }
    #pragma unroll
    for (int t = 0; t < 4; ++t) {
      const int col = t * 16 + ln;
      const int cc  = col < 48 ? col : 48;          // clamp (pad rows/cols
      #pragma unroll
      for (int r = 0; r < 8; ++r) {                 //  are discarded later)
        const int row = wave * 16 + g * 8 + r;
        const int rr  = row < 48 ? row : 48;
        const float mv = MS[rr * 49 + cc];          // unconditional ds_load
        const float l  = (col < 49) ? fmaf(S[t][r], SCALE_, mv) : -1e30f;
        S[t][r] = l;
        mx[r] = fmaxf(mx[r], l);
      }
    }
    #pragma unroll
    for (int r = 0; r < 8; ++r)
      #pragma unroll
      for (int off = 8; off; off >>= 1)
        mx[r] = fmaxf(mx[r], __shfl_xor(mx[r], off, 32)); // 16-lane row reduce
    #pragma unroll
    for (int t = 0; t < 4; ++t)
      #pragma unroll
      for (int r = 0; r < 8; ++r) {
        const float e = __expf(S[t][r] - mx[r]);
        S[t][r] = e; sm[r] += e;
      }
    #pragma unroll
    for (int r = 0; r < 8; ++r) {
      #pragma unroll
      for (int off = 8; off; off >>= 1) sm[r] += __shfl_xor(sm[r], off, 32);
      sm[r] = 1.f / sm[r];
    }

    // ---- P to LDS (C layout -> row-major) ----
    h16* pw = &P[wave * 16 * 72];
    #pragma unroll
    for (int t = 0; t < 4; ++t)
      #pragma unroll
      for (int r = 0; r < 8; ++r)
        pw[(g * 8 + r) * 72 + t * 16 + ln] = (h16)(S[t][r] * sm[r]);

    __syncthreads();  // Vt fully staged for all waves

    // ---- O = P * V : 2 K-steps x 2 N-tiles ----
    v8f O[2];
    O[0] = (v8f){0,0,0,0,0,0,0,0};
    O[1] = (v8f){0,0,0,0,0,0,0,0};
    #pragma unroll
    for (int kt = 0; kt < 2; ++kt) {
      const h16* p = &pw[ln * 72 + kt * 32 + g * 8];
      v16h aP = cat16(*(const v8h*)p, *(const v8h*)(p + 16));
      #pragma unroll
      for (int nt = 0; nt < 2; ++nt) {
        const h16* q = &Vt[(nt * 16 + ln) * 72 + kt * 32 + g * 8];
        v16h bV = cat16(*(const v8h*)q, *(const v8h*)(q + 16));
        O[nt] = wmma_f16(aP, bV, O[nt]);
      }
    }

    #pragma unroll
    for (int nt = 0; nt < 2; ++nt)
      #pragma unroll
      for (int r = 0; r < 8; ++r) {
        const int row = wave * 16 + g * 8 + r;
        const int col = nt * 16 + ln;
        out[((size_t)win * NPAD + row) * C_ + hd0 + col] = (h16)O[nt][r];
      }
  }
}

// ---------------------------------------------------------------------------
// Kernel 5: window reverse + reverse shift + residual + LN2.
// One wave per token: x2 = x + proj_out(mapped);  h2 = LN2(x2) in f16.
// ---------------------------------------------------------------------------
__global__ __launch_bounds__(256)
void merge_ln2_kernel(const float* __restrict__ x,
                      const float* __restrict__ proj,  // [MWIN,192] padded
                      const float* __restrict__ g2,
                      const float* __restrict__ b2,
                      float* __restrict__ x2,
                      h16*   __restrict__ h2) {
  const int tid  = threadIdx.x;
  const int lane = tid & 31;
  const int tok  = blockIdx.x * 8 + (tid >> 5);
  if (tok >= MTOK) return;

  const int b  = tok / (H_ * W_);
  const int r  = tok % (H_ * W_);
  const int hh = r / W_, wc = r % W_;
  const int hs = (hh + H_ - SS_) % H_;   // reverse of roll(+SS)
  const int ws = (wc + W_ - SS_) % W_;
  const int win = b * 64 + (hs / WS_) * 8 + (ws / WS_);
  const int row = (hs % WS_) * WS_ + (ws % WS_);
  const float* ps = proj + ((size_t)win * NPAD + row) * C_;
  const float* xs = x + (size_t)tok * C_;

  float f[6];
  float s = 0.f;
  #pragma unroll
  for (int t = 0; t < 6; ++t) {
    const int c = lane + 32 * t;
    f[t] = xs[c] + ps[c];
    x2[(size_t)tok * C_ + c] = f[t];
    s += f[t];
  }
  #pragma unroll
  for (int off = 16; off; off >>= 1) s += __shfl_xor(s, off, 32);
  const float mean = s * (1.f / C_);
  float v = 0.f;
  #pragma unroll
  for (int t = 0; t < 6; ++t) { float d = f[t] - mean; v += d * d; }
  #pragma unroll
  for (int off = 16; off; off >>= 1) v += __shfl_xor(v, off, 32);
  const float rs = rsqrtf(v * (1.f / C_) + 1e-5f);
  #pragma unroll
  for (int t = 0; t < 6; ++t) {
    const int c = lane + 32 * t;
    h2[(size_t)tok * C_ + c] = (h16)((f[t] - mean) * rs * g2[c] + b2[c]);
  }
}

// ---------------------------------------------------------------------------
// Host-side orchestration (graph-capture safe: only kernel launches).
// Workspace layout (bytes):
//   win_f16  [MWIN*192] f16 :   0        .. 50331648
//   qkv_f16  [MWIN*576] f16 :  50331648  .. 201326592
//   attn_f16 [MWIN*192] f16 : 201326592  .. 251658240
//   proj_f32 [MWIN*192] f32 : 251658240  .. 352321536
//   x2_f32   [MTOK*192] f32 : 352321536  .. 429391872
//   h2_f16   [MTOK*192] f16 : 429391872  .. 467927040
//   mid_f16  [MTOK*768] f16 : 467927040  .. 622067712
// ---------------------------------------------------------------------------
extern "C" void kernel_launch(void* const* d_in, const int* in_sizes, int n_in,
                              void* d_out, int out_size, void* d_ws, size_t ws_size,
                              hipStream_t stream) {
  const float* x       = (const float*)d_in[0];
  const float* norm1_g = (const float*)d_in[1];
  const float* norm1_b = (const float*)d_in[2];
  const float* qkv_w   = (const float*)d_in[3];
  const float* qkv_b   = (const float*)d_in[4];
  const float* proj_w  = (const float*)d_in[5];
  const float* proj_b  = (const float*)d_in[6];
  const float* norm2_g = (const float*)d_in[7];
  const float* norm2_b = (const float*)d_in[8];
  const float* fc1_w   = (const float*)d_in[9];
  const float* fc1_b   = (const float*)d_in[10];
  const float* fc2_w   = (const float*)d_in[11];
  const float* fc2_b   = (const float*)d_in[12];
  const float* amask   = (const float*)d_in[13];

  char* ws = (char*)d_ws;
  h16*   win_f16  = (h16*)  (ws + 0);
  h16*   qkv_f16  = (h16*)  (ws + 50331648ull);
  h16*   attn_f16 = (h16*)  (ws + 201326592ull);
  float* proj_f32 = (float*)(ws + 251658240ull);
  float* x2_f32   = (float*)(ws + 352321536ull);
  h16*   h2_f16   = (h16*)  (ws + 429391872ull);
  h16*   mid_f16  = (h16*)  (ws + 467927040ull);
  float* out      = (float*)d_out;

  // 1. LN1 + shift + window partition (f16, padded to 64 rows/window)
  ln1_window_kernel<<<NWIN, 256, 0, stream>>>(x, norm1_g, norm1_b, win_f16);

  // 2. QKV GEMM: [131072,192] x [192,576]
  gemm_wmma<0><<<dim3(MWIN / BM, 576 / BN), 256, 0, stream>>>(
      win_f16, qkv_w, qkv_b, qkv_f16, nullptr, MWIN, 576, C_);

  // 3. Windowed attention (2048 windows, 6 heads looped per block)
  attn_wmma_kernel<<<NWIN, 128, 0, stream>>>(qkv_f16, amask, attn_f16);

  // 4. Proj GEMM: [131072,192] x [192,192] -> f32 (padded window layout)
  gemm_wmma<1><<<dim3(MWIN / BM, C_ / BN), 256, 0, stream>>>(
      attn_f16, proj_w, proj_b, proj_f32, nullptr, MWIN, C_, C_);

  // 5. Window reverse + residual + LN2
  merge_ln2_kernel<<<MTOK / 8, 256, 0, stream>>>(
      x, proj_f32, norm2_g, norm2_b, x2_f32, h2_f16);

  // 6. FC1 + GELU: [100352,192] x [192,768]
  gemm_wmma<2><<<dim3(MTOK / BM, HID_ / BN), 256, 0, stream>>>(
      h2_f16, fc1_w, fc1_b, mid_f16, nullptr, MTOK, HID_, C_);

  // 7. FC2 + residual -> d_out: [100352,768] x [768,192]
  gemm_wmma<3><<<dim3(MTOK / BM, C_ / BN), 256, 0, stream>>>(
      mid_f16, fc2_w, fc2_b, out, x2_f32, MTOK, C_, HID_);
}
